// MATAPCell_13091060319000
// MI455X (gfx1250) — matrix operand order repaired and checked by reading, not compiled, not run on hardware
//
#include <hip/hip_runtime.h>

// ---------------------------------------------------------------------------
// MATAP cell for gfx1250 (MI455X).
//  - All GEMMs: v_wmma_f32_16x16x32_bf16 (bf16 in, fp32 accumulate), weights
//    pre-packed into the exact B-matrix VGPR layout (pure b128 loads in loop).
//  - Attention is algebraically refactored (qlen=1, softmax weights sum to 1):
//      scores[b,h,m] = mem[b,m,:].p[b,h,:] + q.bk ;  p = q @ blockdiag(Wk^T)
//      context      = wmem @ Wcomb + bias_comb ;     wmem = sum_m attn*mem
//    so K/V (2x80MB) are never materialized and mem is read exactly once.
//  - mem tile staged to LDS with GLOBAL_LOAD_ASYNC_TO_LDS_B128 (ASYNCcnt);
//    the shifted memory output slots 0..8 are emitted from LDS in-place.
// ---------------------------------------------------------------------------

typedef __attribute__((ext_vector_type(16))) __bf16 v16bf;
typedef __attribute__((ext_vector_type(8)))  float  v8f;
typedef __attribute__((ext_vector_type(4)))  float  v4f;

#define ACT_NONE 0
#define ACT_TANH 1
#define ACT_SIG  2

__device__ __forceinline__ v8f wmma_bf16(v16bf a, v16bf b, v8f c) {
  return __builtin_amdgcn_wmma_f32_16x16x32_bf16(false, a, false, b, (short)0, c,
                                                 false, false);
}

// A-matrix 16x32 bf16 layout (wave32):
//  lanes 0-15 : row = m0+lane,    K = k0+{0..7, 16..23}
//  lanes 16-31: row = m0+lane-16, K = k0+{8..15, 24..31}
// kstart must already include k0 + ((lane>>4)*8).
__device__ __forceinline__ v16bf load_a_bf16(const float* __restrict__ A,
                                             long lda, long row, int kstart) {
  const float* p = A + row * lda + kstart;
  v4f f0 = *(const v4f*)(p);
  v4f f1 = *(const v4f*)(p + 4);
  v4f f2 = *(const v4f*)(p + 16);
  v4f f3 = *(const v4f*)(p + 20);
  v16bf a;
#pragma unroll
  for (int i = 0; i < 4; ++i) {
    a[i]      = (__bf16)f0[i];
    a[i + 4]  = (__bf16)f1[i];
    a[i + 8]  = (__bf16)f2[i];
    a[i + 12] = (__bf16)f3[i];
  }
  return a;
}

__device__ __forceinline__ float sigf(float x) {
  return 1.0f / (1.0f + __expf(-x));
}

__device__ __forceinline__ float wave_allred(float v) {
#pragma unroll
  for (int off = 16; off >= 1; off >>= 1) v += __shfl_xor(v, off, 32);
  return v;
}

// ---------------------------------------------------------------------------
// Weight pre-pack: fp32 (K x N, row-major) -> bf16 B-matrix VGPR layout.
// Tile (kt,nt) = 512 bf16; lane l: col l&15, K = kt*32 + (l>=16?8:0) + i + (i>=8?8:0).
// ---------------------------------------------------------------------------
__global__ void pack_w_kernel(const float* __restrict__ W, __bf16* __restrict__ P,
                              int K, int N) {
  int gid  = blockIdx.x * blockDim.x + threadIdx.x;
  int lane = gid & 31;
  int tile = gid >> 5;
  int ntiles = N >> 4;
  int ktiles = K >> 5;
  if (tile >= ktiles * ntiles) return;
  int kt = tile / ntiles, nt = tile % ntiles;
  int c  = lane & 15;
  int kb = kt * 32 + ((lane >> 4) ? 8 : 0);
  __bf16* dst = P + (size_t)tile * 512 + (size_t)lane * 16;
  const float* src = W + (size_t)kb * N + (size_t)nt * 16 + c;
#pragma unroll
  for (int i = 0; i < 16; ++i) dst[i] = (__bf16)src[(size_t)(i + (i >= 8 ? 8 : 0)) * N];
}

// Pack blockdiag(Wk^T): virtual matrix (K=256 rows (h,dk), N=1024 cols (h',d)),
// value = (h==h') ? Wk[d*256 + h*64 + dk] : 0.   Wk is (D=256, H=4, DK=64).
__global__ void pack_wkT_kernel(const float* __restrict__ Wk,
                                __bf16* __restrict__ P) {
  int gid  = blockIdx.x * blockDim.x + threadIdx.x;
  int lane = gid & 31;
  int tile = gid >> 5;               // ktiles=8, ntiles=64 -> 512 tiles
  if (tile >= 512) return;
  int kt = tile >> 6, nt = tile & 63;
  int c  = lane & 15;
  int kb = kt * 32 + ((lane >> 4) ? 8 : 0);
  int cc = nt * 16 + c;              // output col
  int h2 = cc >> 8, d = cc & 255;
  __bf16* dst = P + (size_t)tile * 512 + (size_t)lane * 16;
#pragma unroll
  for (int i = 0; i < 16; ++i) {
    int kk = kb + i + (i >= 8 ? 8 : 0); // (h,dk): h=kk>>6, dk=kk&63
    float v = ((kk >> 6) == h2) ? Wk[(size_t)d * 256 + h2 * 64 + (kk & 63)] : 0.0f;
    dst[i] = (__bf16)v;
  }
}

// Wcomb[(h,d'),d] = sum_dk Wv[d',h,dk] * Wo[h,dk,d]    (1024 x 256, fp32)
__global__ void make_wcomb_kernel(const float* __restrict__ Wv,
                                  const float* __restrict__ Wo,
                                  float* __restrict__ Wcomb) {
  int idx = blockIdx.x * blockDim.x + threadIdx.x;   // 1024*256
  if (idx >= 1024 * 256) return;
  int d = idx & 255, row = idx >> 8;
  int h = row >> 8, dp = row & 255;
  float acc = 0.0f;
#pragma unroll 4
  for (int dk = 0; dk < 64; ++dk)
    acc += Wv[(size_t)dp * 256 + h * 64 + dk] * Wo[(size_t)(h * 64 + dk) * 256 + d];
  Wcomb[idx] = acc;
}

// bias_comb[d] = bo[d] + sum_i bv[i] * Wo[i,d]   (bv flat over (h,dk), 256)
__global__ void make_bias_comb_kernel(const float* __restrict__ bv,
                                      const float* __restrict__ Wo,
                                      const float* __restrict__ bo,
                                      float* __restrict__ bc) {
  int d = blockIdx.x * blockDim.x + threadIdx.x;
  if (d >= 256) return;
  float acc = bo[d];
  for (int i = 0; i < 256; ++i) acc += bv[i] * Wo[(size_t)i * 256 + d];
  bc[d] = acc;
}

// ---------------------------------------------------------------------------
// Generic GEMM: C[rows x N] = act(A[rows x K] @ W + bias); N covered by
// grid.y*256; wave = 16x64 tile (4 wmma accs); two A sources split at Ksplit.
// ---------------------------------------------------------------------------
template <int ACT>
__global__ void gemm_bf16_kernel(const float* __restrict__ A0,
                                 const float* __restrict__ A1, int Ksplit,
                                 int lda, const __bf16* __restrict__ Bp,
                                 int ntiles, const float* __restrict__ bias,
                                 float* __restrict__ C, int ldc, int K) {
  const int lane  = threadIdx.x & 31;
  const int wave  = threadIdx.x >> 5;
  const long m0   = (long)blockIdx.x * 16;
  const int nbase = blockIdx.y * 256 + wave * 64;
  const int khalf = (lane >> 4) * 8;
  const long rowA = m0 + (lane & 15);

  v8f acc[4] = {v8f{}, v8f{}, v8f{}, v8f{}};

  for (int k0 = 0; k0 < K; k0 += 32) {
    const float* A = (k0 < Ksplit) ? A0 : A1;
    const int kk   = (k0 < Ksplit) ? k0 : (k0 - Ksplit);
    __builtin_prefetch(A + rowA * lda + kk + khalf + 32, 0, 1);
    v16bf a = load_a_bf16(A, lda, rowA, kk + khalf);
    const __bf16* bt =
        Bp + ((size_t)(k0 >> 5) * ntiles + (nbase >> 4)) * 512 + (size_t)lane * 16;
#pragma unroll
    for (int t = 0; t < 4; ++t) {
      v16bf b = *(const v16bf*)(bt + (size_t)t * 512);
      acc[t] = wmma_bf16(a, b, acc[t]);
    }
  }

  const int c     = lane & 15;
  const int rhalf = (lane >> 4) * 8;
#pragma unroll
  for (int t = 0; t < 4; ++t) {
    const int ncol = nbase + t * 16 + c;
    const float bv = bias ? bias[ncol] : 0.0f;
#pragma unroll
    for (int j = 0; j < 8; ++j) {
      float x = acc[t][j] + bv;
      if (ACT == ACT_TANH) x = tanhf(x);
      if (ACT == ACT_SIG)  x = sigf(x);
      C[(m0 + rhalf + j) * (long)ldc + ncol] = x;
    }
  }
}

// ---------------------------------------------------------------------------
// Fused GRU (keras reset_after=True). Wave computes a 16x16 slice of gru_out;
// 6 WMMA streams (z/r/h x {kernel, recurrent}).
// ---------------------------------------------------------------------------
__global__ void gru_fused_kernel(const float* __restrict__ zt,
                                 const float* __restrict__ hprev,
                                 const __bf16* __restrict__ Gk,
                                 const __bf16* __restrict__ Grk,
                                 const float* __restrict__ gb,
                                 float* __restrict__ out) {
  const int lane  = threadIdx.x & 31;
  const int wave  = threadIdx.x >> 5;
  const long m0   = (long)blockIdx.x * 16;
  const int n0    = (blockIdx.y * 4 + wave) * 16;
  const int khalf = (lane >> 4) * 8;
  const long rowA = m0 + (lane & 15);

  v8f ax[3] = {v8f{}, v8f{}, v8f{}};
  v8f ah[3] = {v8f{}, v8f{}, v8f{}};

  for (int k0 = 0; k0 < 256; k0 += 32) {
    v16bf a_z = load_a_bf16(zt,    256, rowA, k0 + khalf);
    v16bf a_h = load_a_bf16(hprev, 256, rowA, k0 + khalf);
    const size_t ktbase = (size_t)(k0 >> 5) * 48;  // N=768 -> 48 n-tiles
#pragma unroll
    for (int s = 0; s < 3; ++s) {
      const size_t toff = (ktbase + (size_t)((n0 + s * 256) >> 4)) * 512 +
                          (size_t)lane * 16;
      v16bf b1 = *(const v16bf*)(Gk + toff);
      v16bf b2 = *(const v16bf*)(Grk + toff);
      ax[s] = wmma_bf16(a_z, b1, ax[s]);
      ah[s] = wmma_bf16(a_h, b2, ah[s]);
    }
  }

  const int c     = lane & 15;
  const int col   = n0 + c;
  const int rhalf = (lane >> 4) * 8;
  const float bz0 = gb[col],       br0 = gb[256 + col],  bh0 = gb[512 + col];
  const float bz1 = gb[768 + col], br1 = gb[1024 + col], bh1 = gb[1280 + col];
#pragma unroll
  for (int j = 0; j < 8; ++j) {
    const long row = m0 + rhalf + j;
    const float hp = hprev[row * 256 + col];
    const float z  = sigf(ax[0][j] + bz0 + ah[0][j] + bz1);
    const float r  = sigf(ax[1][j] + br0 + ah[1][j] + br1);
    const float hc = tanhf(ax[2][j] + bh0 + r * (ah[2][j] + bh1));
    out[row * 256 + col] = z * hp + (1.0f - z) * hc;
  }
}

// ---------------------------------------------------------------------------
// Fused attention + memory shift. One workgroup (128 thr = 4 waves) per batch
// row b; wave h handles head h; lane owns an 8-wide d-chunk.
//   - mem[b] (10x256 f32) async-staged to LDS (GLOBAL_LOAD_ASYNC_TO_LDS_B128)
//   - scores_m = (p[b,h,:].mem[b,m,:] + q.bk) / 8 ; softmax over m
//   - wmem[b,h,:] = sum_m attn_m * mem[b,m,:]
//   - emits shifted memory slots 0..8 of the output straight from LDS
// ---------------------------------------------------------------------------
__global__ void __launch_bounds__(128)
attn_fused_kernel(const float* __restrict__ q, const float* __restrict__ p,
                  const float* __restrict__ bk, const float* __restrict__ mem,
                  float* __restrict__ wmem, float* __restrict__ out_mem) {
  __shared__ __align__(16) float smem[2560];
  const int b    = blockIdx.x;
  const int tid  = threadIdx.x;
  const int lane = tid & 31;
  const int h    = tid >> 5;
  const float* gsrc = mem + (size_t)b * 2560;

  // stage 2560 floats: 640 x b128, 5 per thread, via async DMA to LDS
#pragma unroll
  for (int it = 0; it < 5; ++it) {
    const int i = tid + it * 128;
    unsigned lp = (unsigned)(unsigned long long)&smem[i * 4];
    unsigned long long gp = (unsigned long long)(gsrc + i * 4);
    asm volatile("global_load_async_to_lds_b128 %0, %1, off"
                 :: "v"(lp), "v"(gp) : "memory");
  }
  asm volatile("s_wait_asynccnt 0x0" ::: "memory");
  __syncthreads();

  const int dbase = lane * 8;
  const float* pp = p + (size_t)b * 1024 + h * 256 + dbase;
  v4f p0 = *(const v4f*)(pp);
  v4f p1 = *(const v4f*)(pp + 4);

  // q . bk  (64 elems, 2 per lane)
  const float* qp = q + (size_t)b * 256 + h * 64;
  const float* bkp = bk + h * 64;
  float tpart = qp[2 * lane] * bkp[2 * lane] + qp[2 * lane + 1] * bkp[2 * lane + 1];
  const float t = wave_allred(tpart);

  float w[10];
#pragma unroll
  for (int m = 0; m < 10; ++m) {
    v4f s0 = *(const v4f*)&smem[m * 256 + dbase];
    v4f s1 = *(const v4f*)&smem[m * 256 + dbase + 4];
    float part = p0[0] * s0[0] + p0[1] * s0[1] + p0[2] * s0[2] + p0[3] * s0[3] +
                 p1[0] * s1[0] + p1[1] * s1[1] + p1[2] * s1[2] + p1[3] * s1[3];
    w[m] = (wave_allred(part) + t) * 0.125f;
  }
  float smax = w[0];
#pragma unroll
  for (int m = 1; m < 10; ++m) smax = fmaxf(smax, w[m]);
  float ssum = 0.0f;
#pragma unroll
  for (int m = 0; m < 10; ++m) { w[m] = __expf(w[m] - smax); ssum += w[m]; }
  const float inv = 1.0f / ssum;

  v4f a0 = v4f{}, a1 = v4f{};
#pragma unroll
  for (int m = 0; m < 10; ++m) {
    const float wm = w[m] * inv;
    v4f s0 = *(const v4f*)&smem[m * 256 + dbase];
    v4f s1 = *(const v4f*)&smem[m * 256 + dbase + 4];
    a0 += wm * s0;
    a1 += wm * s1;
  }
  float* wp = wmem + (size_t)b * 1024 + h * 256 + dbase;
  *(v4f*)(wp)     = a0;
  *(v4f*)(wp + 4) = a1;

  // memory shift: out_mem[b, 0:9, :] = smem[256:2560]
  const v4f* sv = (const v4f*)smem;
  v4f* ov = (v4f*)out_mem + (size_t)b * 640;
  for (int i = tid; i < 576; i += 128) ov[i] = sv[64 + i];
}

// ---------------------------------------------------------------------------
// Row LayerNorm over 256 cols: one wave32 per row.
// ---------------------------------------------------------------------------
__device__ __forceinline__ void ln_reduce_write(float* v, int lane, long row,
                                                const float* __restrict__ g,
                                                const float* __restrict__ bta,
                                                float* __restrict__ Y) {
  float s = 0.0f, ss = 0.0f;
#pragma unroll
  for (int i = 0; i < 8; ++i) { s += v[i]; ss += v[i] * v[i]; }
#pragma unroll
  for (int off = 16; off >= 1; off >>= 1) {
    s  += __shfl_xor(s, off, 32);
    ss += __shfl_xor(ss, off, 32);
  }
  const float mu  = s * (1.0f / 256.0f);
  const float var = ss * (1.0f / 256.0f) - mu * mu;
  const float rs  = rsqrtf(var + 1e-3f);
  const int cb = lane * 8;
#pragma unroll
  for (int i = 0; i < 8; ++i)
    Y[row * 256 + cb + i] = (v[i] - mu) * rs * g[cb + i] + bta[cb + i];
}

__global__ void rowln_kernel(const float* __restrict__ X,
                             const float* __restrict__ g,
                             const float* __restrict__ bta,
                             float* __restrict__ Y, int rows) {
  const int wid  = (blockIdx.x * blockDim.x + threadIdx.x) >> 5;
  const int lane = threadIdx.x & 31;
  if (wid >= rows) return;
  const float* x = X + (size_t)wid * 256 + lane * 8;
  float v[8];
#pragma unroll
  for (int i = 0; i < 8; ++i) v[i] = x[i];
  ln_reduce_write(v, lane, wid, g, bta, Y);
}

__global__ void gate_ln_kernel(const float* __restrict__ alpha,
                               const float* __restrict__ gru,
                               const float* __restrict__ cp,
                               const float* __restrict__ g,
                               const float* __restrict__ bta,
                               float* __restrict__ Y, int rows) {
  const int wid  = (blockIdx.x * blockDim.x + threadIdx.x) >> 5;
  const int lane = threadIdx.x & 31;
  if (wid >= rows) return;
  const size_t base = (size_t)wid * 256 + lane * 8;
  float v[8];
#pragma unroll
  for (int i = 0; i < 8; ++i) {
    const float al = alpha[base + i];
    v[i] = (1.0f - al) * gru[base + i] + al * cp[base + i];
  }
  ln_reduce_write(v, lane, wid, g, bta, Y);
}

// ---------------------------------------------------------------------------
// Host-side orchestration
// ---------------------------------------------------------------------------
extern "C" void kernel_launch(void* const* d_in, const int* in_sizes, int n_in,
                              void* d_out, int out_size, void* d_ws,
                              size_t ws_size, hipStream_t stream) {
  constexpr int B = 8192, D = 256, M = 10;
  const float* inputs    = (const float*)d_in[0];
  const float* h_prev    = (const float*)d_in[1];
  const float* mem       = (const float*)d_in[2];
  const float* W_in      = (const float*)d_in[3];
  const float* b_in      = (const float*)d_in[4];
  const float* gru_k     = (const float*)d_in[5];
  const float* gru_rk    = (const float*)d_in[6];
  const float* gru_b     = (const float*)d_in[7];
  const float* Wq        = (const float*)d_in[8];
  const float* bq        = (const float*)d_in[9];
  const float* Wk        = (const float*)d_in[10];
  const float* bk        = (const float*)d_in[11];
  const float* Wv        = (const float*)d_in[12];
  const float* bv        = (const float*)d_in[13];
  const float* Wo        = (const float*)d_in[14];
  const float* bo        = (const float*)d_in[15];
  const float* g_attn    = (const float*)d_in[16];
  const float* beta_attn = (const float*)d_in[17];
  const float* g_out     = (const float*)d_in[18];
  const float* beta_out  = (const float*)d_in[19];
  const float* W_ctx     = (const float*)d_in[20];
  const float* b_ctx     = (const float*)d_in[21];
  const float* W_gate    = (const float*)d_in[22];
  const float* b_gate    = (const float*)d_in[23];
  const float* W_mem     = (const float*)d_in[24];
  const float* b_mem     = (const float*)d_in[25];
  (void)in_sizes; (void)n_in; (void)out_size; (void)ws_size;

  char* ws = (char*)d_ws;
  size_t off = 0;
  auto carve = [&](size_t bytes) -> void* {
    void* pp = ws + off;
    off += (bytes + 255) & ~(size_t)255;
    return pp;
  };

  // packed bf16 weights (~2.5 MB)
  __bf16* wp_in   = (__bf16*)carve((size_t)D * D * 2);
  __bf16* wp_gk   = (__bf16*)carve((size_t)D * 768 * 2);
  __bf16* wp_grk  = (__bf16*)carve((size_t)D * 768 * 2);
  __bf16* wp_q    = (__bf16*)carve((size_t)D * D * 2);
  __bf16* wp_kT   = (__bf16*)carve((size_t)D * 1024 * 2);
  __bf16* wp_comb = (__bf16*)carve((size_t)1024 * D * 2);
  __bf16* wp_ctx  = (__bf16*)carve((size_t)D * D * 2);
  __bf16* wp_gate = (__bf16*)carve((size_t)512 * D * 2);
  __bf16* wp_mem  = (__bf16*)carve((size_t)D * D * 2);
  // fp32 derived weights + intermediates
  float* wcomb_f  = (float*)carve((size_t)1024 * D * 4);
  float* bias_cmb = (float*)carve((size_t)D * 4);
  float* zt       = (float*)carve((size_t)B * D * 4);
  float* gruo     = (float*)carve((size_t)B * D * 4);
  float* qbuf     = (float*)carve((size_t)B * D * 4);
  float* pbuf     = (float*)carve((size_t)B * 1024 * 4);
  float* wmemb    = (float*)carve((size_t)B * 1024 * 4);
  float* ctxraw   = (float*)carve((size_t)B * D * 4);
  float* ctxn     = (float*)carve((size_t)B * D * 4);
  float* ctxp     = (float*)carve((size_t)B * D * 4);
  float* alphab   = (float*)carve((size_t)B * D * 4);

  // derived weights
  make_wcomb_kernel<<<(1024 * 256 + 255) / 256, 256, 0, stream>>>(Wv, Wo, wcomb_f);
  make_bias_comb_kernel<<<1, 256, 0, stream>>>(bv, Wo, bo, bias_cmb);

  auto pack = [&](const float* src, __bf16* dst, int K, int N) {
    int threads = (K >> 5) * (N >> 4) * 32;
    pack_w_kernel<<<(threads + 255) / 256, 256, 0, stream>>>(src, dst, K, N);
  };
  pack(W_in,    wp_in,   256, 256);
  pack(gru_k,   wp_gk,   256, 768);
  pack(gru_rk,  wp_grk,  256, 768);
  pack(Wq,      wp_q,    256, 256);
  pack(W_ctx,   wp_ctx,  256, 256);
  pack(W_gate,  wp_gate, 512, 256);
  pack(W_mem,   wp_mem,  256, 256);
  pack(wcomb_f, wp_comb, 1024, 256);
  pack_wkT_kernel<<<(512 * 32 + 255) / 256, 256, 0, stream>>>(Wk, wp_kT);

  auto gemm = [&](int act, const float* A0, const float* A1, int Ksplit,
                  int lda, int K, const __bf16* Bp, int N, const float* bias,
                  float* C, int ldc, int rows) {
    dim3 g(rows / 16, N / 256), blk(128);
    if (act == ACT_NONE)
      gemm_bf16_kernel<ACT_NONE><<<g, blk, 0, stream>>>(A0, A1, Ksplit, lda, Bp,
                                                        N / 16, bias, C, ldc, K);
    else if (act == ACT_TANH)
      gemm_bf16_kernel<ACT_TANH><<<g, blk, 0, stream>>>(A0, A1, Ksplit, lda, Bp,
                                                        N / 16, bias, C, ldc, K);
    else
      gemm_bf16_kernel<ACT_SIG><<<g, blk, 0, stream>>>(A0, A1, Ksplit, lda, Bp,
                                                       N / 16, bias, C, ldc, K);
  };

  // 1) z_t = inputs @ W_in + b_in
  gemm(ACT_NONE, inputs, inputs, 256, 256, 256, wp_in, 256, b_in, zt, 256, B);
  // 2) GRU -> gru_out
  gru_fused_kernel<<<dim3(B / 16, 4), 128, 0, stream>>>(zt, h_prev, wp_gk,
                                                        wp_grk, gru_b, gruo);
  // 3) q = gru_out @ Wq + bq ;  p = q @ blockdiag(Wk^T)   (B x 1024)
  gemm(ACT_NONE, gruo, gruo, 256, 256, 256, wp_q, 256, bq, qbuf, 256, B);
  gemm(ACT_NONE, qbuf, qbuf, 256, 256, 256, wp_kT, 1024, nullptr, pbuf, 1024, B);
  // 4) fused attention + memory shift (slots 0..8 of output memory)
  float* out_h   = (float*)d_out;
  float* out_mem = out_h + (size_t)B * D;
  attn_fused_kernel<<<B, 128, 0, stream>>>(qbuf, pbuf, bk, mem, wmemb, out_mem);
  // 5) context = LN(wmem @ Wcomb + bias_comb)
  gemm(ACT_NONE, wmemb, wmemb, 1024, 1024, 1024, wp_comb, 256, bias_cmb,
       ctxraw, 256, B);
  rowln_kernel<<<B / 8, 256, 0, stream>>>(ctxraw, g_attn, beta_attn, ctxn, B);
  // 6) ctx_p = tanh(context @ W_ctx + b_ctx)
  gemm(ACT_TANH, ctxn, ctxn, 256, 256, 256, wp_ctx, 256, b_ctx, ctxp, 256, B);
  // 7) alpha = sigmoid([gru_out | ctx_p] @ W_gate + b_gate)
  gemm(ACT_SIG, gruo, ctxp, 256, 256, 512, wp_gate, 256, b_gate, alphab, 256, B);
  // 8) h_corr = LN((1-a)*gru_out + a*ctx_p) -> d_out[0 : B*D]
  gate_ln_kernel<<<B / 8, 256, 0, stream>>>(alphab, gruo, ctxp, g_out,
                                            beta_out, out_h, B);
  // 9) new_entry -> output memory slot 9
  gemm(ACT_NONE, zt, zt, 256, 256, 256, wp_mem, 256, b_mem,
       out_mem + 9 * 256, M * D, B);
}
